// WGM_Module_51479478010027
// MI455X (gfx1250) — matrix operand order repaired and checked by reading, not compiled, not run on hardware
//
#include <hip/hip_runtime.h>
#include <math.h>

typedef __attribute__((ext_vector_type(16))) _Float16 v16h;
typedef __attribute__((ext_vector_type(8)))  float    v8f;

#define B_   4
#define C_   64
#define H_   64
#define W_   64
#define HW_  4096
#define D2_  128
#define K_   4
#define N_   16
#define R_   4
#define MP_  144   // K_*36 projection rows

// ---------------------------------------------------------------------------
// WMMA GEMM: Out[b,m,n] = sum_k Wt[m,k] * In[b,k,n]  (+bias, +act, +scale*resid)
// Templated on K (64/128) and NT N-tiles per wave:
//  - A fragments (weights) loaded once per wave via float4 and reused across NT tiles
//  - KDIM/32 * NT static v_wmma_f32_16x16x32_f16 per wave, f32 accumulate
// ---------------------------------------------------------------------------
template<int KDIM, int NT>
__global__ void gemm_wmma(const float* __restrict__ Wt,    // (M,KDIM) row-major
                          const float* __restrict__ In,    // (B,KDIM,Np)
                          float* __restrict__ Out,         // (B,M,Np)
                          const float* __restrict__ bias,  // (M) or null
                          const float* __restrict__ resid, // (B,M,Np) or null
                          const float* __restrict__ scale, // 1 elem or null
                          int M, int Np, int act)
{
    constexpr int KS = KDIM / 32;
    const int lane = threadIdx.x & 31;
    const int half = lane >> 4;          // 0 or 1 (lane group)
    const int l16  = lane & 15;
    const int ntile0 = blockIdx.x * NT, mtile = blockIdx.y, b = blockIdx.z;
    const int m = mtile * 16 + l16;      // A-matrix row for this lane
    const float* inb  = In + (size_t)b * KDIM * Np;
    const float* wrow = Wt + (size_t)m * KDIM;

    // A 16x32 f16 layout per k-step: lanes0-15 hold K {0..7}(v0-3) + {16..23}(v4-7);
    // lanes16-31 hold K {8..15} + {24..31}.  Two contiguous 8-float runs -> float4 x4.
    v16h a[KS];
    #pragma unroll
    for (int ks = 0; ks < KS; ++ks) {
        const float4* p0 = (const float4*)(wrow + ks * 32 + half * 8);
        const float4* p1 = (const float4*)(wrow + ks * 32 + 16 + half * 8);
        float4 x0 = p0[0], x1 = p0[1], x2 = p1[0], x3 = p1[1];
        a[ks][0]  = (_Float16)x0.x; a[ks][1]  = (_Float16)x0.y;
        a[ks][2]  = (_Float16)x0.z; a[ks][3]  = (_Float16)x0.w;
        a[ks][4]  = (_Float16)x1.x; a[ks][5]  = (_Float16)x1.y;
        a[ks][6]  = (_Float16)x1.z; a[ks][7]  = (_Float16)x1.w;
        a[ks][8]  = (_Float16)x2.x; a[ks][9]  = (_Float16)x2.y;
        a[ks][10] = (_Float16)x2.z; a[ks][11] = (_Float16)x2.w;
        a[ks][12] = (_Float16)x3.x; a[ks][13] = (_Float16)x3.y;
        a[ks][14] = (_Float16)x3.z; a[ks][15] = (_Float16)x3.w;
    }

    v8f acc[NT];
    #pragma unroll
    for (int nt = 0; nt < NT; ++nt) acc[nt] = (v8f){0.f,0.f,0.f,0.f,0.f,0.f,0.f,0.f};

    #pragma unroll
    for (int ks = 0; ks < KS; ++ks) {
        #pragma unroll
        for (int nt = 0; nt < NT; ++nt) {
            const int n = (ntile0 + nt) * 16 + l16;
            // B 32x16 f16 layout: lane = n; lanes0-15 K {0..7}+{16..23}, lanes16-31 K {8..15}+{24..31}.
            v16h bf;
            #pragma unroll
            for (int j = 0; j < 16; ++j)
                bf[j] = (_Float16)inb[(size_t)(ks * 32 + half * 8 + (j & 7) + ((j >> 3) << 4)) * Np + n];
            acc[nt] = __builtin_amdgcn_wmma_f32_16x16x32_f16(false, a[ks], false, bf,
                                                             (short)0, acc[nt], false, false);
        }
    }

    const float s = scale ? scale[0] : 0.f;
    // C/D layout: vgpr r -> m = r + 8*half, n = lane&15
    #pragma unroll
    for (int nt = 0; nt < NT; ++nt) {
        const int n = (ntile0 + nt) * 16 + l16;
        #pragma unroll
        for (int r = 0; r < 8; ++r) {
            int mr = mtile * 16 + r + half * 8;
            float v = acc[nt][r];
            if (bias)  v += bias[mr];
            if (act)   v = v / (1.f + expf(-v));   // SiLU
            size_t oi = ((size_t)b * M + mr) * Np + n;
            if (resid) v += s * resid[oi];
            Out[oi] = v;
        }
    }
}

static inline void launch_gemm(int K, const float* Wt, const float* In, float* Out,
                               const float* bias, const float* resid, const float* scale,
                               int M, int Np, int act, hipStream_t s)
{
    dim3 grid(Np / 64, M / 16, B_);   // NT=4 -> 64 columns per wave
    if (K == 64)
        gemm_wmma<64, 4><<<grid, 32, 0, s>>>(Wt, In, Out, bias, resid, scale, M, Np, act);
    else
        gemm_wmma<128, 4><<<grid, 32, 0, s>>>(Wt, In, Out, bias, resid, scale, M, Np, act);
}

// ---------------------------------------------------------------------------
// Channel LayerNorm over C at each (b,p). In-place safe (one thread per column).
// ---------------------------------------------------------------------------
__global__ void ln_kernel(const float* __restrict__ in, float* __restrict__ out,
                          const float* __restrict__ g, const float* __restrict__ bt,
                          int B, int C, int Np)
{
    int idx = blockIdx.x * blockDim.x + threadIdx.x;
    if (idx >= B * Np) return;
    int b = idx / Np, p = idx % Np;
    const float* base = in + (size_t)b * C * Np + p;
    float s = 0.f, s2 = 0.f;
    for (int c = 0; c < C; ++c) { float x = base[(size_t)c * Np]; s += x; s2 += x * x; }
    float m   = s / C;
    float var = s2 / C - m * m;
    float inv = rsqrtf(var + 1e-6f);
    float* ob = out + (size_t)b * C * Np + p;
    for (int c = 0; c < C; ++c) {
        float x = base[(size_t)c * Np];
        ob[(size_t)c * Np] = (x - m) * inv * g[c] + bt[c];
    }
}

// ---------------------------------------------------------------------------
// Haar DWT: q (B,16,64,64) -> out (B,64,32,32)  [ll|hl|lh|hh channel blocks]
// ---------------------------------------------------------------------------
__global__ void haar_kernel(const float* __restrict__ q, float* __restrict__ out)
{
    int idx = blockIdx.x * blockDim.x + threadIdx.x;
    if (idx >= B_ * 16 * 32 * 32) return;
    int j = idx & 31, i = (idx >> 5) & 31, c = (idx >> 10) & 15, b = idx >> 14;
    const float* qb = q + ((size_t)b * 16 + c) * HW_;
    float a  = qb[(2*i)   * 64 + 2*j];
    float bb = qb[(2*i)   * 64 + 2*j + 1];
    float cc = qb[(2*i+1) * 64 + 2*j];
    float d  = qb[(2*i+1) * 64 + 2*j + 1];
    size_t ob = (size_t)b * 64 * 1024 + (size_t)i * 32 + j;
    out[ob + (size_t)(c)      * 1024] = (a + bb + cc + d) * 0.5f;
    out[ob + (size_t)(16 + c) * 1024] = (bb - a + d - cc) * 0.5f;
    out[ob + (size_t)(32 + c) * 1024] = (cc - a + d - bb) * 0.5f;
    out[ob + (size_t)(48 + c) * 1024] = (a - bb - cc + d) * 0.5f;
}

// wt_ctw (c,o,p,q) -> W2[(o*4+p*2+q), c]  (256 x 64)
__global__ void repack_ctw_kernel(const float* __restrict__ w, float* __restrict__ w2)
{
    int idx = blockIdx.x * blockDim.x + threadIdx.x;
    if (idx >= 256 * 64) return;
    int c = idx & 63, row = idx >> 6;
    int o = row >> 2, p = (row >> 1) & 1, q = row & 1;
    w2[(size_t)row * 64 + c] = w[(((size_t)c * 64 + o) * 2 + p) * 2 + q];
}

// P2 (B,256,1024) -> up (B,64,64,64): up[b,o,2i+p,2j+q] = P2[b,(o,p,q),(i,j)] + ctb[o]
__global__ void upscatter_kernel(const float* __restrict__ p2, const float* __restrict__ ctb,
                                 float* __restrict__ up)
{
    int idx = blockIdx.x * blockDim.x + threadIdx.x;
    if (idx >= B_ * 256 * 1024) return;
    int pos = idx & 1023, m = (idx >> 10) & 255, b = idx >> 18;
    int o = m >> 2, p = (m >> 1) & 1, q = m & 1;
    int i = pos >> 5, j = pos & 31;
    up[(((size_t)b * 64 + o) * 64 + (2*i + p)) * 64 + (2*j + q)] = p2[idx] + ctb[o];
}

// ---------------------------------------------------------------------------
// Generic depthwise conv (square kernel, dilation, optional pre-add + bias + SiLU)
// ---------------------------------------------------------------------------
__global__ void dw_conv_kernel(const float* __restrict__ in, long long inBs,
                               const float* __restrict__ addin, long long addBs,
                               const float* __restrict__ wt,
                               const float* __restrict__ bias,
                               float* __restrict__ out, long long outBs,
                               int B, int C, int ks, int pad, int dil, int act)
{
    int idx = blockIdx.x * blockDim.x + threadIdx.x;
    if (idx >= B * C * HW_) return;
    int p = idx % HW_, c = (idx / HW_) % C, b = idx / (HW_ * C);
    int h = p >> 6, w = p & 63;
    const float* ib = in + (size_t)b * inBs + (size_t)c * HW_;
    const float* ab = addin ? addin + (size_t)b * addBs + (size_t)c * HW_ : nullptr;
    float acc = bias ? bias[c] : 0.f;
    for (int ky = 0; ky < ks; ++ky) {
        int yy = h - pad + ky * dil;
        if ((unsigned)yy >= 64u) continue;
        for (int kx = 0; kx < ks; ++kx) {
            int xx = w - pad + kx * dil;
            if ((unsigned)xx >= 64u) continue;
            float v = ib[yy * 64 + xx];
            if (ab) v += ab[yy * 64 + xx];
            acc += v * wt[(c * ks + ky) * ks + kx];
        }
    }
    if (act) acc = acc / (1.f + expf(-acc));
    out[(size_t)b * outBs + (size_t)c * HW_ + p] = acc;
}

// ---------------------------------------------------------------------------
// SS2D selective scan. One lane per (b,k,d,n); 16-lane groups per d-channel.
// Scan-order flips/transposes handled via index remap into row-major X/P.
// ---------------------------------------------------------------------------
__global__ void ss2d_scan_kernel(const float* __restrict__ X,    // (B,D2,HW)
                                 const float* __restrict__ P,    // (B,144,HW)
                                 const float* __restrict__ dtw,  // (K,D2,R)
                                 const float* __restrict__ dtb,  // (K,D2)
                                 const float* __restrict__ alog, // (K,D2,N)
                                 float* __restrict__ ys)         // (B,K,D2,HW)
{
    const int tid = blockIdx.x * blockDim.x + threadIdx.x;
    const int row = tid >> 4;                // (b,k,d)
    const int n   = tid & 15;
    const int d   = row % D2_;
    const int k   = (row / D2_) & 3;
    const int b   = row / (D2_ * 4);

    const int kd  = k * D2_ + d;
    const float an = -expf(alog[(size_t)kd * N_ + n]);
    const float w0 = dtw[(size_t)kd * 4 + 0];
    const float w1 = dtw[(size_t)kd * 4 + 1];
    const float w2 = dtw[(size_t)kd * 4 + 2];
    const float w3 = dtw[(size_t)kd * 4 + 3];
    const float bb = dtb[kd];

    const float* Xr = X + ((size_t)b * D2_ + d) * HW_;
    const float* Pb = P + (size_t)b * MP_ * HW_ + (size_t)(k * 36) * HW_;
    float* yrow = ys + ((size_t)(b * 4 + k) * D2_ + d) * HW_;

    float hstate = 0.f;
    for (int l = 0; l < HW_; ++l) {
        int pos;
        if      (k == 0) pos = l;
        else if (k == 1) pos = ((l & 63) << 6) | (l >> 6);            // transposed order
        else if (k == 2) pos = HW_ - 1 - l;                           // reversed
        else { int lt = HW_ - 1 - l; pos = ((lt & 63) << 6) | (lt >> 6); }

        float u    = Xr[pos];
        float draw = bb + w0 * Pb[pos] + w1 * Pb[HW_ + pos]
                        + w2 * Pb[2 * HW_ + pos] + w3 * Pb[3 * HW_ + pos];
        float dt   = (draw > 20.f) ? draw : log1pf(expf(draw));       // softplus
        float Bn   = Pb[(size_t)(4 + n)  * HW_ + pos];
        float Cn   = Pb[(size_t)(20 + n) * HW_ + pos];

        hstate = hstate * expf(dt * an) + dt * u * Bn;
        float contrib = hstate * Cn;
        contrib += __shfl_xor(contrib, 1, 32);
        contrib += __shfl_xor(contrib, 2, 32);
        contrib += __shfl_xor(contrib, 4, 32);
        contrib += __shfl_xor(contrib, 8, 32);
        if (n == 0) yrow[l] = contrib;
    }
}

// Merge 4 directions + D*u skip term: out (B,D2,HW)
__global__ void ss_combine_kernel(const float* __restrict__ ys, const float* __restrict__ X,
                                  const float* __restrict__ dp, float* __restrict__ out)
{
    int idx = blockIdx.x * blockDim.x + threadIdx.x;
    if (idx >= B_ * D2_ * HW_) return;
    int p = idx % HW_, d = (idx / HW_) % D2_, b = idx / (HW_ * D2_);
    int h = p >> 6, w = p & 63;
    int pt = (w << 6) | h;
    size_t base = ((size_t)b * 4 * D2_ + d) * HW_;
    size_t sk   = (size_t)D2_ * HW_;
    float o = ys[base + p]
            + ys[base + 2 * sk + (HW_ - 1 - p)]
            + ys[base + sk + pt]
            + ys[base + 3 * sk + (HW_ - 1 - pt)];
    float sdp = dp[d] + dp[D2_ + d] + dp[2 * D2_ + d] + dp[3 * D2_ + d];
    out[idx] = o + sdp * X[idx];
}

// ---------------------------------------------------------------------------
// ECA: spatial mean per (b,c), 1D conv over channels, sigmoid gate, apply.
// ---------------------------------------------------------------------------
__global__ void mean_hw_kernel(const float* __restrict__ in, float* __restrict__ v)
{
    __shared__ float sm[256];
    const float* base = in + (size_t)blockIdx.x * HW_;
    float s = 0.f;
    for (int p = threadIdx.x; p < HW_; p += 256) s += base[p];
    sm[threadIdx.x] = s;
    __syncthreads();
    for (int off = 128; off; off >>= 1) {
        if (threadIdx.x < (unsigned)off) sm[threadIdx.x] += sm[threadIdx.x + off];
        __syncthreads();
    }
    if (threadIdx.x == 0) v[blockIdx.x] = sm[0] * (1.f / HW_);
}

__global__ void eca_gate_kernel(const float* __restrict__ v, const float* __restrict__ w,
                                float* __restrict__ g)
{
    int i = threadIdx.x;
    if (i >= B_ * C_) return;
    int c = i % C_;
    float left  = (c > 0)       ? v[i - 1] : 0.f;
    float mid   = v[i];
    float right = (c < C_ - 1)  ? v[i + 1] : 0.f;
    float z = w[0] * left + w[1] * mid + w[2] * right;
    g[i] = 1.f / (1.f + expf(-z));
}

__global__ void gate_apply_kernel(const float* __restrict__ q, const float* __restrict__ g,
                                  const float* __restrict__ x, const float* __restrict__ wgm,
                                  float* __restrict__ out)
{
    int idx = blockIdx.x * blockDim.x + threadIdx.x;
    if (idx >= B_ * C_ * HW_) return;
    int c = (idx / HW_) % C_, b = idx / (HW_ * C_);
    out[idx] = g[b * C_ + c] * q[idx] + wgm[0] * x[idx];
}

// ---------------------------------------------------------------------------
// Host-side orchestration
// ---------------------------------------------------------------------------
extern "C" void kernel_launch(void* const* d_in, const int* in_sizes, int n_in,
                              void* d_out, int out_size, void* d_ws, size_t ws_size,
                              hipStream_t stream)
{
    (void)in_sizes; (void)n_in; (void)out_size; (void)ws_size;
    const float* x       = (const float*)d_in[0];
    const float* wt_dw   = (const float*)d_in[1];
    const float* wt_db   = (const float*)d_in[2];
    const float* wt_lng  = (const float*)d_in[3];
    const float* wt_lnb  = (const float*)d_in[4];
    const float* wt_ctw  = (const float*)d_in[5];
    const float* wt_ctb  = (const float*)d_in[6];
    const float* cv_inw  = (const float*)d_in[7];
    const float* cv_ln1g = (const float*)d_in[8];
    const float* cv_ln1b = (const float*)d_in[9];
    const float* cv_pw   = (const float*)d_in[10];
    const float* cv_dww  = (const float*)d_in[11];
    const float* cv_dwb  = (const float*)d_in[12];
    const float* ss_xp   = (const float*)d_in[13];   // (4,36,128) == (144,128) row-major
    const float* ss_dtw  = (const float*)d_in[14];
    const float* ss_dtb  = (const float*)d_in[15];
    const float* ss_alog = (const float*)d_in[16];
    const float* ss_d    = (const float*)d_in[17];
    const float* cv_ong  = (const float*)d_in[18];
    const float* cv_onb  = (const float*)d_in[19];
    const float* cv_ow   = (const float*)d_in[20];
    const float* cv_skip = (const float*)d_in[21];
    const float* eca_w   = (const float*)d_in[22];
    const float* wgm_skip= (const float*)d_in[23];
    const float* pf_nw   = (const float*)d_in[24];
    const float* pf_nb   = (const float*)d_in[25];
    const float* pf_lng  = (const float*)d_in[26];
    const float* pf_lnb  = (const float*)d_in[27];
    const float* pf_d1   = (const float*)d_in[28];
    const float* pf_d2   = (const float*)d_in[29];
    const float* pf_d3   = (const float*)d_in[30];
    const float* pf_d4   = (const float*)d_in[31];
    const float* pf_flng = (const float*)d_in[32];
    const float* pf_flnb = (const float*)d_in[33];
    const float* pf_fw   = (const float*)d_in[34];
    const float* pf_fb   = (const float*)d_in[35];
    const float* pf_skip = (const float*)d_in[36];
    float* out = (float*)d_out;

    // Arena (floats), manually overlapped by lifetime. Peak ~72 MB.
    float* A0 = (float*)d_ws;            // 8,388,608  : q1 -> P2 -> ys
    float* A1 = A0 + 8388608;            // 2,359,296  : qcat -> up -> P -> xg -> t5
    float* A2 = A1 + 2359296;            // 2,097,152  : W2 -> X -> cat
    float* A3 = A2 + 2097152;            // 2,097,152  : h128a -> ssout
    float* A4 = A3 + 2097152;            // 1,048,576  : h64 (skip_in, long-lived)
    float* A5 = A4 + 1048576;            // 1,048,576  : qbuf
    float* A6 = A5 + 1048576;            // 1,048,576  : res (long-lived)
    float* SS = A6 + 1048576;            // 1,024      : eca means + gates

    const long long BS64  = 64LL  * HW_;
    const long long BS128 = 128LL * HW_;

    // 1) q = c1(x, wt_dw) + wt_db ; LN  -> A0 (B,16,HW)
    launch_gemm(64, wt_dw, x, A0, wt_db, nullptr, nullptr, 16, HW_, 0, stream);
    ln_kernel<<<(B_*HW_+255)/256, 256, 0, stream>>>(A0, A0, wt_lng, wt_lnb, B_, 16, HW_);

    // 2) Haar DWT -> qcat A1 (B,64,1024)
    haar_kernel<<<(B_*16*1024+255)/256, 256, 0, stream>>>(A0, A1);

    // 3) transposed conv 2x2 s2 as GEMM: repack W -> A2; P2 -> A0; scatter+bias -> up A1
    repack_ctw_kernel<<<(256*64+255)/256, 256, 0, stream>>>(wt_ctw, A2);
    launch_gemm(64, A2, A1, A0, nullptr, nullptr, nullptr, 256, 1024, 0, stream);
    upscatter_kernel<<<(B_*256*1024+255)/256, 256, 0, stream>>>(A0, wt_ctb, A1);

    // 4) h = LN(c1(up, cv_inw)) -> A4 (skip_in)
    launch_gemm(64, cv_inw, A1, A4, nullptr, nullptr, nullptr, 64, HW_, 0, stream);
    ln_kernel<<<(B_*HW_+255)/256, 256, 0, stream>>>(A4, A4, cv_ln1g, cv_ln1b, B_, 64, HW_);

    // 5) h128 = c1(h, cv_pw) -> A3 ; X = silu(dw3x3(h128)+b) -> A2
    launch_gemm(64, cv_pw, A4, A3, nullptr, nullptr, nullptr, 128, HW_, 0, stream);
    dw_conv_kernel<<<(B_*128*HW_+255)/256, 256, 0, stream>>>(A3, BS128, nullptr, 0, cv_dww, cv_dwb, A2, BS128, B_, 128, 3, 1, 1, 1);

    // 6) SS2D: projection GEMM (144x128) -> P A1 ; scan -> ys A0 ; merge -> A3 ; LN
    launch_gemm(128, ss_xp, A2, A1, nullptr, nullptr, nullptr, MP_, HW_, 0, stream);
    ss2d_scan_kernel<<<128, 256, 0, stream>>>(A2, A1, ss_dtw, ss_dtb, ss_alog, A0);
    ss_combine_kernel<<<(B_*D2_*HW_+255)/256, 256, 0, stream>>>(A0, A2, ss_d, A3);
    ln_kernel<<<(B_*HW_+255)/256, 256, 0, stream>>>(A3, A3, cv_ong, cv_onb, B_, 128, HW_);

    // 7) q = c1(h, cv_ow) + cv_skip * skip_in -> A5
    launch_gemm(128, cv_ow, A3, A5, nullptr, A4, cv_skip, 64, HW_, 0, stream);

    // 8) ECA gate + global skip -> xg A1
    mean_hw_kernel<<<B_*C_, 256, 0, stream>>>(A5, SS);
    eca_gate_kernel<<<1, 256, 0, stream>>>(SS, eca_w, SS + 512);
    gate_apply_kernel<<<(B_*C_*HW_+255)/256, 256, 0, stream>>>(A5, SS + 512, x, wgm_skip, A1);

    // 9) res = LN(c1(xg, pf_nw)+pf_nb) -> A6
    launch_gemm(64, pf_nw, A1, A6, pf_nb, nullptr, nullptr, 64, HW_, 0, stream);
    ln_kernel<<<(B_*HW_+255)/256, 256, 0, stream>>>(A6, A6, pf_lng, pf_lnb, B_, 64, HW_);

    // 10) chained depthwise convs into cat A2 (channel blocks of 16)
    dw_conv_kernel<<<(B_*16*HW_+255)/256, 256, 0, stream>>>(A6,          BS64, nullptr,     0,    pf_d1, nullptr, A2,          BS64, B_, 16, 3, 1, 1, 0);
    dw_conv_kernel<<<(B_*16*HW_+255)/256, 256, 0, stream>>>(A2,          BS64, A6 + 16*HW_, BS64, pf_d2, nullptr, A2 + 16*HW_, BS64, B_, 16, 7, 3, 1, 0);
    dw_conv_kernel<<<(B_*16*HW_+255)/256, 256, 0, stream>>>(A2 + 16*HW_, BS64, A6 + 32*HW_, BS64, pf_d3, nullptr, A2 + 32*HW_, BS64, B_, 16, 3, 2, 2, 0);
    dw_conv_kernel<<<(B_*16*HW_+255)/256, 256, 0, stream>>>(A2 + 32*HW_, BS64, A6 + 48*HW_, BS64, pf_d4, nullptr, A2 + 48*HW_, BS64, B_, 16, 3, 3, 3, 0);

    // 11) out = silu(c1(LN(cat), pf_fw)+pf_fb) + pf_skip * res
    ln_kernel<<<(B_*HW_+255)/256, 256, 0, stream>>>(A2, A1, pf_flng, pf_flnb, B_, 64, HW_);
    launch_gemm(64, pf_fw, A1, out, pf_fb, A6, pf_skip, 64, HW_, 1, stream);
}
